// TopoAlignGRU_54752243089493
// MI455X (gfx1250) — compile-verified, matrix operand order checked by reading, not run on hardware
//
#include <hip/hip_runtime.h>
#include <hip/hip_bf16.h>
#include <stdint.h>

// ---------------- model constants (from reference) ----------------
#define N_BUS   33
#define BATCH   32
#define WSTEPS  192
#define BW      (BATCH*WSTEPS)     // 6144
#define ROWS    (BW*N_BUS)         // 202752
#define GNNH    128
#define HID     256
#define MEAS    64
#define FEAT    32

typedef unsigned short u16;
typedef __attribute__((ext_vector_type(16))) __bf16 v16bf;
typedef __attribute__((ext_vector_type(8)))  float  v8f;
typedef __attribute__((ext_vector_type(4)))  unsigned int u32x4;
typedef __attribute__((ext_vector_type(8)))  int i32x8;
typedef __attribute__((ext_vector_type(4)))  int i32x4;

union Frag { uint4 q[2]; v16bf v; };

__device__ __forceinline__ float bf2f(u16 h){ return __uint_as_float(((uint32_t)h)<<16); }
__device__ __forceinline__ u16   f2bf(float f){
  uint32_t u = __float_as_uint(f);
  u += 0x7fffu + ((u>>16)&1u);           // round-to-nearest-even
  return (u16)(u>>16);
}
__device__ __forceinline__ float sigm(float x){ return 1.f/(1.f+__expf(-x)); }

// ---------------- Tensor Data Mover availability ----------------
#if defined(__has_builtin)
# if __has_builtin(__builtin_amdgcn_tensor_load_to_lds)
#  define USE_TDM 1
# else
#  define USE_TDM 0
# endif
#else
# define USE_TDM 0
#endif

#if USE_TDM
// DMA one 64-row x 32-col bf16 tile (row stride = lda elements) from global into
// LDS at byte offset ldsOff, inserting 16B padding after every 64B row
// (pad_interval code 3 = 16 DWORDs, pad_amount code 3 = 4 DWORDs) -> 80B LDS rows.
__device__ __forceinline__ void tdm_load_tile(unsigned ldsOff, const u16* gptr, unsigned ldaElems){
  unsigned long long ga = (unsigned long long)(uintptr_t)gptr;
  const unsigned dim0 = 1u<<30, dim1 = 1u<<30;   // huge: no OOB clipping
  u32x4 g0;
  g0[0] = 1u;                                    // count=1, user descriptor
  g0[1] = ldsOff;                                // lds_addr
  g0[2] = (unsigned)ga;                          // global_addr[31:0]
  g0[3] = (unsigned)((ga>>32) & 0x1FFFFFFu) | (2u<<30);  // global_addr[56:32] | type=2
  i32x8 g1;
  g1[0] = (int)((1u<<16) | (1u<<20) | (3u<<22) | (3u<<25)); // data_size=2B, pad_en, int=16DW, amt=4DW
  g1[1] = (int)((dim0 & 0xFFFFu) << 16);                    // tensor_dim0[15:0]
  g1[2] = (int)(((dim0>>16)&0xFFFFu) | ((dim1&0xFFFFu)<<16));
  g1[3] = (int)(((dim1>>16)&0xFFFFu) | (32u<<16));          // tile_dim0 = 32 elems
  g1[4] = 64;                                               // tile_dim1 = 64 rows, tile_dim2 = 0
  g1[5] = (int)ldaElems;                                    // tensor_dim0_stride lo
  g1[6] = 0;
  g1[7] = 0;
  i32x4 g2 = {0,0,0,0}, g3 = {0,0,0,0};
#if __clang_major__ >= 23
  i32x8 g4 = {0,0,0,0,0,0,0,0};
  __builtin_amdgcn_tensor_load_to_lds(g0, g1, g2, g3, g4, 0);
#else
  __builtin_amdgcn_tensor_load_to_lds(g0, g1, g2, g3, 0);
#endif
}
__device__ __forceinline__ unsigned ldsOffOf(const void* p){ return (unsigned)(uintptr_t)p; }
#endif

// ---------------- fp32 -> bf16 weight conversion ----------------
__global__ void k_cvt(const float* __restrict__ s, u16* __restrict__ d, int n){
  for(int i=blockIdx.x*blockDim.x+threadIdx.x;i<n;i+=gridDim.x*blockDim.x) d[i]=f2bf(s[i]);
}

// ---------------- graph prep: norm, P^1..P^3, rowsums ----------------
__global__ void k_prep(const int* __restrict__ eidx, const float* __restrict__ ew, int E,
                       float* __restrict__ P, float* __restrict__ R){
  __shared__ float deg[N_BUS], dinv[N_BUS];
  __shared__ float P1[N_BUS*N_BUS], P2[N_BUS*N_BUS], P3[N_BUS*N_BUS];
  int t = threadIdx.x;
  if (t<N_BUS) deg[t]=0.f;
  for(int i=t;i<N_BUS*N_BUS;i+=blockDim.x) P1[i]=0.f;
  __syncthreads();
  if(t==0){ for(int e=0;e<E;++e) deg[eidx[E+e]] += ew[e]; }
  __syncthreads();
  if(t<N_BUS) dinv[t] = deg[t]>0.f ? rsqrtf(deg[t]) : 0.f;
  __syncthreads();
  if(t==0){
    for(int e=0;e<E;++e){ int s=eidx[e], d=eidx[E+e]; P1[d*N_BUS+s] += dinv[s]*ew[e]*dinv[d]; }
  }
  __syncthreads();
  for(int i=t;i<N_BUS*N_BUS;i+=blockDim.x){
    int r=i/N_BUS,c=i%N_BUS; float a=0.f;
    for(int s=0;s<N_BUS;++s) a+=P1[r*N_BUS+s]*P1[s*N_BUS+c];
    P2[i]=a;
  }
  __syncthreads();
  for(int i=t;i<N_BUS*N_BUS;i+=blockDim.x){
    int r=i/N_BUS,c=i%N_BUS; float a=0.f;
    for(int s=0;s<N_BUS;++s) a+=P2[r*N_BUS+s]*P1[s*N_BUS+c];
    P3[i]=a;
  }
  __syncthreads();
  for(int i=t;i<N_BUS*N_BUS;i+=blockDim.x){ P[i]=P1[i]; P[1089+i]=P2[i]; P[2178+i]=P3[i]; }
  if(t<N_BUS){
    float a=0,b=0,c=0;
    for(int s=0;s<N_BUS;++s){ a+=P1[t*N_BUS+s]; b+=P2[t*N_BUS+s]; c+=P3[t*N_BUS+s]; }
    R[t]=a; R[N_BUS+t]=b; R[2*N_BUS+t]=c;
  }
}

// ---------------- pack [z|feat] -> bf16 (6144 x 96) ----------------
__global__ void k_pack(const float* __restrict__ z, const float* __restrict__ f, u16* __restrict__ s){
  int n = BW*(MEAS+FEAT);
  for(int i=blockIdx.x*blockDim.x+threadIdx.x;i<n;i+=gridDim.x*blockDim.x){
    int r=i/(MEAS+FEAT), c=i%(MEAS+FEAT);
    float v = (c<MEAS)? z[(size_t)r*MEAS+c] : f[(size_t)r*FEAT + (c-MEAS)];
    s[i]=f2bf(v);
  }
}

// ---------------- generic bf16 WMMA GEMM: C = act(A(MxK) * B(NxK)^T + bias) ----------------
// block = 128 threads (4 waves), tile 64x64, K-step 32, fp32 accumulate.
// TDM path: double-buffered LDS, wave 0 issues tensor DMA, one barrier per K-step.
__global__ __launch_bounds__(128)
void k_gemm(const u16* __restrict__ A, int lda,
            const u16* __restrict__ Bm, int ldb,
            const float* __restrict__ bias,
            u16* __restrict__ Cb, float* __restrict__ Cf,
            int N, int K, int act){
  __shared__ __align__(16) u16 sA[2][64*40];   // +8 bf16 pad per row
  __shared__ __align__(16) u16 sB[2][64*40];
  const int tid  = threadIdx.x;
  const int lane = tid & 31;
  const int wave = tid >> 5;
  const int wm = (wave>>1)&1, wn = wave&1;
  const int m0 = blockIdx.x*64, n0 = blockIdx.y*64;
  const int lh = lane & 15;
  const int kb = (lane<16)?0:8;               // ISA 16-bit A/B lane K-offset pattern

  v8f acc[2][2];
  #pragma unroll
  for(int i=0;i<2;++i){ acc[i][0]=0; acc[i][1]=0; }

#if USE_TDM
  const u16* gA = A  + (size_t)m0*lda;
  const u16* gB = Bm + (size_t)n0*ldb;
  const unsigned offA0 = ldsOffOf(&sA[0][0]), offA1 = ldsOffOf(&sA[1][0]);
  const unsigned offB0 = ldsOffOf(&sB[0][0]), offB1 = ldsOffOf(&sB[1][0]);
  if (wave==0){
    tdm_load_tile(offA0, gA, (unsigned)lda);
    tdm_load_tile(offB0, gB, (unsigned)ldb);
  }
  int buf = 0;
  for(int k0=0;k0<K;k0+=32){
    if (wave==0) __builtin_amdgcn_s_wait_tensorcnt(0);
    __syncthreads();                          // tiles in buf are valid for all waves
    if (wave==0 && k0+32<K){                  // prefetch next K-slab into other buffer
      tdm_load_tile(buf? offA0:offA1, gA + k0 + 32, (unsigned)lda);
      tdm_load_tile(buf? offB0:offB1, gB + k0 + 32, (unsigned)ldb);
    }
    const u16* bufA = sA[buf];
    const u16* bufB = sB[buf];
    Frag af[2], bf[2];
    #pragma unroll
    for(int i=0;i<2;++i){
      int ra = (wm*32 + i*16 + lh)*40 + kb;
      af[i].q[0]=*(const uint4*)&bufA[ra]; af[i].q[1]=*(const uint4*)&bufA[ra+16];
      int rb = (wn*32 + i*16 + lh)*40 + kb;
      bf[i].q[0]=*(const uint4*)&bufB[rb]; bf[i].q[1]=*(const uint4*)&bufB[rb+16];
    }
    #pragma unroll
    for(int i=0;i<2;++i)
      #pragma unroll
      for(int j=0;j<2;++j)
        acc[i][j] = __builtin_amdgcn_wmma_f32_16x16x32_bf16(
            false, af[i].v, false, bf[j].v, (short)0, acc[i][j], false, false);
    buf ^= 1;
  }
#else
  const int lr = tid>>1, lc = (tid&1)*16;     // loader: 64 rows x 32 cols, 16 bf16/thread
  const u16* ga = A  + (size_t)(m0+lr)*lda + lc;
  const u16* gb = Bm + (size_t)(n0+lr)*ldb + lc;
  for(int k0=0;k0<K;k0+=32){
    uint4 ra0 = *(const uint4*)(ga+k0);
    uint4 ra1 = *(const uint4*)(ga+k0+8);
    uint4 rb0 = *(const uint4*)(gb+k0);
    uint4 rb1 = *(const uint4*)(gb+k0+8);
    *(uint4*)&sA[0][lr*40+lc]   = ra0;
    *(uint4*)&sA[0][lr*40+lc+8] = ra1;
    *(uint4*)&sB[0][lr*40+lc]   = rb0;
    *(uint4*)&sB[0][lr*40+lc+8] = rb1;
    __syncthreads();
    if (k0+32<K){ __builtin_prefetch(ga+k0+32,0,0); __builtin_prefetch(gb+k0+32,0,0); }
    Frag af[2], bf[2];
    #pragma unroll
    for(int i=0;i<2;++i){
      int ra = (wm*32 + i*16 + lh)*40 + kb;
      af[i].q[0]=*(uint4*)&sA[0][ra]; af[i].q[1]=*(uint4*)&sA[0][ra+16];
      int rb = (wn*32 + i*16 + lh)*40 + kb;
      bf[i].q[0]=*(uint4*)&sB[0][rb]; bf[i].q[1]=*(uint4*)&sB[0][rb+16];
    }
    #pragma unroll
    for(int i=0;i<2;++i)
      #pragma unroll
      for(int j=0;j<2;++j)
        acc[i][j] = __builtin_amdgcn_wmma_f32_16x16x32_bf16(
            false, af[i].v, false, bf[j].v, (short)0, acc[i][j], false, false);
    __syncthreads();
  }
#endif

  const int rBase = m0 + wm*32 + ((lane<16)?0:8);
  const int cBase = n0 + wn*32 + lh;
  #pragma unroll
  for(int i=0;i<2;++i){
    #pragma unroll
    for(int j=0;j<2;++j){
      int col = cBase + j*16;
      float bs = bias ? bias[col] : 0.f;
      #pragma unroll
      for(int v=0; v<8; ++v){
        float x = acc[i][j][v] + bs;
        if (act) x = fmaxf(x, 0.f);
        size_t idx = (size_t)(rBase + i*16 + v)*N + col;
        if (Cf) Cf[idx]=x; else Cb[idx]=f2bf(x);
      }
    }
  }
}

// ---------------- TAG layer 0 combine: x1 = relu(y0 + r_k[n]*y_k + b) ----------------
__global__ void k_combine(const u16* __restrict__ y, const float* __restrict__ R,
                          const float* __restrict__ b0, u16* __restrict__ x1){
  int total = ROWS*GNNH;
  for(int i=blockIdx.x*blockDim.x+threadIdx.x;i<total;i+=gridDim.x*blockDim.x){
    int f = i & (GNNH-1);
    int rn = i >> 7;
    int n  = rn % N_BUS;
    int bw = rn / N_BUS;
    const u16* yr = y + (size_t)bw*(4*GNNH);
    float v = bf2f(yr[f])
            + R[n]         * bf2f(yr[GNNH+f])
            + R[N_BUS+n]   * bf2f(yr[2*GNNH+f])
            + R[2*N_BUS+n] * bf2f(yr[3*GNNH+f])
            + b0[f];
    x1[i] = f2bf(fmaxf(v,0.f));
  }
}

// ---------------- TAG layer 1 banded mix: out1 = Z0 + sum_k P^k . Z_k + b ----------------
__global__ void k_mix(const u16* __restrict__ Z, const float* __restrict__ P,
                      const float* __restrict__ b1, u16* __restrict__ out1){
  int total = ROWS*GNNH;
  for(int i=blockIdx.x*blockDim.x+threadIdx.x;i<total;i+=gridDim.x*blockDim.x){
    int f = i & (GNNH-1);
    int row = i >> 7;
    int n  = row % N_BUS;
    int bw = row / N_BUS;
    float acc = bf2f(Z[(size_t)row*(4*GNNH) + f]) + b1[f];
    #pragma unroll
    for(int ds=-3; ds<=3; ++ds){
      int s = n + ds;
      if (s < 0 || s >= N_BUS) continue;
      size_t zr = ((size_t)bw*N_BUS + s)*(4*GNNH);
      int a = ds<0 ? -ds : ds; if (a<1) a=1;
      for(int k=a;k<=3;++k)
        acc += P[(k-1)*1089 + n*N_BUS + s] * bf2f(Z[zr + k*GNNH + f]);
    }
    out1[i]=f2bf(acc);
  }
}

// ---------------- gated fusion: u = a*g~ + (1-a)*s~, a = sigmoid(A1 + t2) (u aliases g~) ----
__global__ void k_fuse(const u16* __restrict__ A1, const float* __restrict__ t2,
                       const u16* __restrict__ gt, const u16* __restrict__ stil,
                       u16* __restrict__ u){
  int total = ROWS*HID;
  for(int i=blockIdx.x*blockDim.x+threadIdx.x;i<total;i+=gridDim.x*blockDim.x){
    int c = i & (HID-1);
    int row = i >> 8;
    int bw = row / N_BUS;
    float al = sigm(bf2f(A1[i]) + t2[(size_t)bw*HID + c]);
    float g  = bf2f(gt[i]);
    float s  = bf2f(stil[(size_t)bw*HID + c]);
    u[i] = f2bf(al*g + (1.f-al)*s);
  }
}

// ---------------- decoder head: per-row 2 dots of 256, wave-parallel ----------------
__global__ __launch_bounds__(256)
void k_head(const u16* __restrict__ hdec, const u16* __restrict__ w2,
            const float* __restrict__ b2, float* __restrict__ out){
  int wave = threadIdx.x >> 5, lane = threadIdx.x & 31;
  int row = blockIdx.x*8 + wave;
  if (row >= ROWS) return;
  int j0 = lane*8;
  const u16* hr = hdec + (size_t)row*HID + j0;
  const u16* wa = w2 + j0;          // d2_w row 0 -> va
  const u16* wm = w2 + HID + j0;    // d2_w row 1 -> vm
  float s0=0.f, s1=0.f;
  #pragma unroll
  for(int u=0;u<8;++u){ float h=bf2f(hr[u]); s0 += h*bf2f(wa[u]); s1 += h*bf2f(wm[u]); }
  for(int off=16; off>0; off>>=1){ s0 += __shfl_down(s0,off,32); s1 += __shfl_down(s1,off,32); }
  if(lane==0){
    int bw = row / N_BUS, n = row % N_BUS;
    out[(size_t)bw*(2*N_BUS) + n]         = s1 + b2[1];   // vm
    out[(size_t)bw*(2*N_BUS) + N_BUS + n] = s0 + b2[0];   // va
  }
}

// ---------------- persistent single-WG GRU layer (WMMA recurrent matmul) ----------------
// gi = x@wih^T + bih precomputed (fp32, rows b*W+t, 768 cols). h lives in LDS (bf16 for WMMA).
__global__ __launch_bounds__(256)
void k_gru(const float* __restrict__ gi, const u16* __restrict__ whh,
           const float* __restrict__ bhh, u16* __restrict__ xt){
  __shared__ __align__(16) u16 hb[BATCH*264];    // bf16 h, padded rows
  __shared__ float ghc[BATCH*264];               // one 256-col chunk of gh (fp32)
  const int tid = threadIdx.x;
  const int wave = tid >> 5, lane = tid & 31;
  const int lh = lane & 15;
  const int kb = (lane<16)?0:8;
  const int b  = tid >> 3;            // 0..31 batch row owned for gate math
  const int c0 = (tid & 7) * 32;      // 32 columns owned within [0,256)

  float hv[32];
  #pragma unroll
  for(int j=0;j<32;++j) hv[j]=0.f;
  for(int i=tid;i<BATCH*264;i+=blockDim.x) hb[i]=0;
  __syncthreads();

  float rg[3][32];

  for(int t=0;t<WSTEPS;++t){
    // gh = h(32x256) @ whh^T(768x256), in 3 chunks of 256 output cols
    for(int ch=0; ch<3; ++ch){
      #pragma unroll
      for(int q=0;q<4;++q){
        int idx = wave*4 + q;            // 0..31 frag within chunk
        int mt  = idx >> 4;              // 0..1
        int ntl = idx & 15;              // 0..15
        v8f c = 0;
        #pragma unroll
        for(int k0=0;k0<HID;k0+=32){
          Frag fa, fb;
          int ra = (mt*16 + lh)*264 + k0 + kb;
          fa.q[0]=*(uint4*)&hb[ra]; fa.q[1]=*(uint4*)&hb[ra+16];
          const u16* gB = whh + (size_t)(ch*HID + ntl*16 + lh)*HID + k0 + kb;
          fb.q[0]=*(const uint4*)gB; fb.q[1]=*(const uint4*)(gB+16);
          c = __builtin_amdgcn_wmma_f32_16x16x32_bf16(false,fa.v,false,fb.v,(short)0,c,false,false);
        }
        int rBase = mt*16 + ((lane<16)?0:8);
        int col   = ntl*16 + lh;
        #pragma unroll
        for(int v=0;v<8;++v) ghc[(rBase+v)*264 + col] = c[v];
      }
      __syncthreads();
      #pragma unroll
      for(int j=0;j<32;++j) rg[ch][j] = ghc[b*264 + c0 + j];
      __syncthreads();
    }
    // gates
    size_t grow = ((size_t)b*WSTEPS + t)*(3*HID);
    size_t orow = ((size_t)b*WSTEPS + t)*HID;
    #pragma unroll
    for(int j=0;j<32;++j){
      int c = c0 + j;
      float ir  = gi[grow + c];
      float iz  = gi[grow + HID + c];
      float inn = gi[grow + 2*HID + c];
      float r = sigm(ir + rg[0][j] + bhh[c]);
      float z = sigm(iz + rg[1][j] + bhh[HID+c]);
      float nn = tanhf(inn + r*(rg[2][j] + bhh[2*HID+c]));
      float hnew = (1.f - z)*nn + z*hv[j];
      hv[j] = hnew;
      hb[b*264 + c] = f2bf(hnew);
      xt[orow + c] = f2bf(hnew);
    }
    __syncthreads();
  }
}

// =======================================================================
extern "C" void kernel_launch(void* const* d_in, const int* in_sizes, int n_in,
                              void* d_out, int out_size, void* d_ws, size_t ws_size,
                              hipStream_t stream){
  (void)n_in; (void)out_size; (void)ws_size;
  const float* z_seq = (const float*)d_in[0];
  const float* feat  = (const float*)d_in[1];
  const int*   eidx  = (const int*)d_in[2];
  const float* ew    = (const float*)d_in[3];
  const int E = in_sizes[3];
  const float* prm[30];
  for(int i=0;i<30;++i) prm[i]=(const float*)d_in[4+i];
  enum {PRE_W=0,PRE_B,TAG0_W,TAG0_B,TAG1_W,TAG1_B,POST_W,POST_B,TP1_W,TP1_B,TP2_W,TP2_B,
        G0_WIH,G0_WHH,G0_BIH,G0_BHH,G1_WIH,G1_WHH,G1_BIH,G1_BHH,AG_W,AG_B,AS_W,AS_B,
        GATE_W,GATE_B,D1_W,D1_B,D2_W,D2_B};

  char* ws = (char*)d_ws;
  size_t cur = 0;
  auto alloc = [&](size_t bytes)->char*{
    char* r = ws + cur; cur += (bytes + 255) & ~(size_t)255; return r;
  };

  // persistent small
  float* P = (float*)alloc(3*1089*sizeof(float));
  float* R = (float*)alloc(3*N_BUS*sizeof(float));
  // bf16 weights
  u16* wb_pre  = (u16*)alloc((size_t)GNNH*FEAT*2);
  u16* wb_tag0 = (u16*)alloc((size_t)4*GNNH*GNNH*2);
  u16* wb_tag1 = (u16*)alloc((size_t)4*GNNH*GNNH*2);
  u16* wb_post = (u16*)alloc((size_t)HID*GNNH*2);
  u16* wb_tp1  = (u16*)alloc((size_t)HID*(MEAS+FEAT)*2);
  u16* wb_tp2  = (u16*)alloc((size_t)HID*HID*2);
  u16* wb_g0i  = (u16*)alloc((size_t)3*HID*HID*2);
  u16* wb_g0h  = (u16*)alloc((size_t)3*HID*HID*2);
  u16* wb_g1i  = (u16*)alloc((size_t)3*HID*HID*2);
  u16* wb_g1h  = (u16*)alloc((size_t)3*HID*HID*2);
  u16* wb_ag   = (u16*)alloc((size_t)HID*HID*2);
  u16* wb_as   = (u16*)alloc((size_t)HID*HID*2);
  u16* wb_gate = (u16*)alloc((size_t)HID*2*HID*2);
  u16* wb_d1   = (u16*)alloc((size_t)HID*HID*2);
  u16* wb_d2   = (u16*)alloc((size_t)2*HID*2);
  // temporal activations
  u16* s_in = (u16*)alloc((size_t)BW*(MEAS+FEAT)*2);
  u16* s1   = (u16*)alloc((size_t)BW*HID*2);
  u16* s2   = (u16*)alloc((size_t)BW*HID*2);
  float* gi = (float*)alloc((size_t)BW*3*HID*sizeof(float));  // reused for both GRU layers
  u16* xt0  = (u16*)alloc((size_t)BW*HID*2);
  u16* st   = (u16*)alloc((size_t)BW*HID*2);
  u16* stil = (u16*)alloc((size_t)BW*HID*2);
  float* t2 = (float*)alloc((size_t)BW*HID*sizeof(float));
  // spatial activations (with reuse)
  u16* xpre = (u16*)alloc((size_t)BW*GNNH*2);
  u16* y    = (u16*)alloc((size_t)BW*4*GNNH*2);
  u16* x1s  = (u16*)alloc((size_t)ROWS*GNNH*2);        // x1, later out1
  u16* zsl  = (u16*)alloc((size_t)ROWS*4*GNNH*2);      // Z; later g(+0) and A1(+half); later hdec(+0)
  u16* gts  = (u16*)alloc((size_t)ROWS*HID*2);         // g_tilde, then u (in place)

  u16* out1 = x1s;
  u16* g    = zsl;
  u16* A1   = zsl + (size_t)ROWS*HID;
  u16* hdec = zsl;
  u16* u_   = gts;

  auto cvt = [&](const float* src, u16* dst, int n){
    k_cvt<<<dim3((n+255)/256), dim3(256), 0, stream>>>(src, dst, n);
  };
  auto gemm = [&](const u16* A, int lda, const u16* Bm, int ldb, const float* bias,
                  u16* Cb, float* Cf, int M, int N, int K, int act){
    k_gemm<<<dim3(M/64, N/64), dim3(128), 0, stream>>>(A, lda, Bm, ldb, bias, Cb, Cf, N, K, act);
  };

  // ---- prep: graph coefficients + bf16 weights ----
  k_prep<<<dim3(1), dim3(128), 0, stream>>>(eidx, ew, E, P, R);
  cvt(prm[PRE_W],  wb_pre,  GNNH*FEAT);
  cvt(prm[TAG0_W], wb_tag0, 4*GNNH*GNNH);
  cvt(prm[TAG1_W], wb_tag1, 4*GNNH*GNNH);
  cvt(prm[POST_W], wb_post, HID*GNNH);
  cvt(prm[TP1_W],  wb_tp1,  HID*(MEAS+FEAT));
  cvt(prm[TP2_W],  wb_tp2,  HID*HID);
  cvt(prm[G0_WIH], wb_g0i,  3*HID*HID);
  cvt(prm[G0_WHH], wb_g0h,  3*HID*HID);
  cvt(prm[G1_WIH], wb_g1i,  3*HID*HID);
  cvt(prm[G1_WHH], wb_g1h,  3*HID*HID);
  cvt(prm[AG_W],   wb_ag,   HID*HID);
  cvt(prm[AS_W],   wb_as,   HID*HID);
  cvt(prm[GATE_W], wb_gate, HID*2*HID);
  cvt(prm[D1_W],   wb_d1,   HID*HID);
  cvt(prm[D2_W],   wb_d2,   2*HID);

  // ---- temporal encoder ----
  k_pack<<<dim3((BW*(MEAS+FEAT)+255)/256), dim3(256), 0, stream>>>(z_seq, feat, s_in);
  gemm(s_in, MEAS+FEAT, wb_tp1, MEAS+FEAT, prm[TP1_B], s1, nullptr, BW, HID, MEAS+FEAT, 1);
  gemm(s1,   HID,       wb_tp2, HID,       prm[TP2_B], s2, nullptr, BW, HID, HID, 1);
  gemm(s2,   HID, wb_g0i, HID, prm[G0_BIH], nullptr, gi, BW, 3*HID, HID, 0);
  k_gru<<<dim3(1), dim3(256), 0, stream>>>(gi, wb_g0h, prm[G0_BHH], xt0);
  gemm(xt0,  HID, wb_g1i, HID, prm[G1_BIH], nullptr, gi, BW, 3*HID, HID, 0);
  k_gru<<<dim3(1), dim3(256), 0, stream>>>(gi, wb_g1h, prm[G1_BHH], st);
  gemm(st,   HID, wb_as,  HID, prm[AS_B],   stil, nullptr, BW, HID, HID, 0);
  gemm(stil, HID, wb_gate + HID, 2*HID, prm[GATE_B], nullptr, t2, BW, HID, HID, 0);

  // ---- spatial encoder ----
  gemm(s_in + MEAS, MEAS+FEAT, wb_pre, FEAT, prm[PRE_B], xpre, nullptr, BW, GNNH, FEAT, 0);
  gemm(xpre, GNNH, wb_tag0, GNNH, nullptr, y, nullptr, BW, 4*GNNH, GNNH, 0);
  k_combine<<<dim3((ROWS*GNNH+255)/256), dim3(256), 0, stream>>>(y, R, prm[TAG0_B], x1s);
  gemm(x1s, GNNH, wb_tag1, GNNH, nullptr, zsl, nullptr, ROWS, 4*GNNH, GNNH, 0);
  k_mix<<<dim3((ROWS*GNNH+255)/256), dim3(256), 0, stream>>>(zsl, P, prm[TAG1_B], out1);
  gemm(out1, GNNH, wb_post, GNNH, prm[POST_B], g, nullptr, ROWS, HID, GNNH, 0);

  // ---- gated fusion + decoder ----
  gemm(g,   HID, wb_ag,   HID,   prm[AG_B], gts, nullptr, ROWS, HID, HID, 0);
  gemm(gts, HID, wb_gate, 2*HID, nullptr,   A1,  nullptr, ROWS, HID, HID, 0);
  k_fuse<<<dim3((ROWS*HID+255)/256), dim3(256), 0, stream>>>(A1, t2, gts, stil, u_);
  gemm(u_,  HID, wb_d1,   HID,   prm[D1_B], hdec, nullptr, ROWS, HID, HID, 1);
  k_head<<<dim3((ROWS+7)/8), dim3(256), 0, stream>>>(hdec, wb_d2, prm[D2_B], (float*)d_out);
}
